// MultiheadAttention_50053548868006
// MI455X (gfx1250) — compile-verified, matrix operand order checked by reading, not run on hardware
//
#include <hip/hip_runtime.h>
#include <hip/hip_bf16.h>

// ---------------------------------------------------------------------------
// Multi-head attention forward for MI455X (gfx1250, wave32, WMMA + TDM).
//   B=2, S=2048, D_MODEL=1024, H=16, d=64.
//   - f16 WMMA everywhere, f32 accumulate
//   - TDM (tensor_load_to_lds) tile staging with descriptor-side LDS padding
//     for bank-conflict-free ds_load_b128 fragment reads
//   - double-buffered LDS in GEMM and attention, one barrier per step
//   - V pre-transposed to [B][H][d][S] so attention tiles are pure TDM loads
//   - exp2-based online softmax (log2e folded into the q scale)
// ---------------------------------------------------------------------------

typedef _Float16 half8  __attribute__((ext_vector_type(8)));
typedef _Float16 half16 __attribute__((ext_vector_type(16)));
typedef float    float8 __attribute__((ext_vector_type(8)));
typedef unsigned int uint4v __attribute__((ext_vector_type(4)));
typedef unsigned int uint8v __attribute__((ext_vector_type(8)));

#define USE_TDM 1

// Padded LDS row strides (halves): conflict-free bank mapping for 16-row frags.
//   GEMM:      32 data + 8 pad = 40  (r*20 mod 64 all distinct)
//   attention: 64 data + 8 pad = 72  (r*36 mod 64 all distinct)
#define GEMM_LDSW 40
#define ATTN_LDSW 72

// Pack-convert 8 fp32 -> 8 f16 with v_cvt_pk_rtz_f16_f32.
__device__ __forceinline__ half8 cvt8(const float* __restrict__ src) {
    float4 a = *(const float4*)(src);
    float4 b = *(const float4*)(src + 4);
    auto p0 = __builtin_amdgcn_cvt_pkrtz(a.x, a.y);
    auto p1 = __builtin_amdgcn_cvt_pkrtz(a.z, a.w);
    auto p2 = __builtin_amdgcn_cvt_pkrtz(b.x, b.y);
    auto p3 = __builtin_amdgcn_cvt_pkrtz(b.z, b.w);
    union { decltype(p0) h2[4]; half8 v; } u;
    u.h2[0] = p0; u.h2[1] = p1; u.h2[2] = p2; u.h2[3] = p3;
    return u.v;
}

// Row-major 16x32 f16 WMMA fragment load (CDNA5 16-bit A/B layout).
__device__ __forceinline__ half16 load_frag(const _Float16* tile, int stride, int lane) {
    const int r  = lane & 15;
    const int kh = (lane >> 4) << 3;      // 0 or 8
    const _Float16* p = tile + r * stride + kh;
    half8 lo = *(const half8*)(p);
    half8 hi = *(const half8*)(p + 16);
    return __builtin_shufflevector(lo, hi, 0,1,2,3,4,5,6,7,8,9,10,11,12,13,14,15);
}

#if USE_TDM
// TDM 2D tile load: global (row-major, stride_d0 elems/row) -> LDS, optional
// LDS-side padding (pad_interval/pad_amount codes per CDNA5 ISA ch.8).
__device__ __forceinline__ void tdm_load_2d(const void* gsrc, unsigned lds_addr,
                                            unsigned tensor_d0, unsigned tensor_d1,
                                            unsigned tile_d0, unsigned tile_d1,
                                            unsigned stride_d0,
                                            unsigned pad_i, unsigned pad_a, bool pad_en)
{
    const unsigned long long ga = (unsigned long long)gsrc;
    uint4v g0;
    g0[0] = 1u;                                                // count=1 (valid user D#)
    g0[1] = lds_addr;
    g0[2] = (unsigned)(ga & 0xFFFFFFFFu);                      // global_addr[31:0]
    g0[3] = (unsigned)((ga >> 32) & 0x01FFFFFFu) | (2u << 30); // addr[56:32] | type=2
    uint8v g1;
    g1[0] = (1u << 16)                                         // data_size = 2 bytes
          | (pad_en ? (1u << 20) : 0u)                         // pad_enable
          | ((pad_i & 7u) << 22)                               // pad_interval code
          | ((pad_a & 127u) << 25);                            // pad_amount code
    g1[1] = (tensor_d0 & 0xFFFFu) << 16;
    g1[2] = ((tensor_d0 >> 16) & 0xFFFFu) | ((tensor_d1 & 0xFFFFu) << 16);
    g1[3] = ((tensor_d1 >> 16) & 0xFFFFu) | ((tile_d0 & 0xFFFFu) << 16);
    g1[4] = tile_d1 & 0xFFFFu;                                 // tile_dim2 = 0
    g1[5] = stride_d0;
    g1[6] = 0u;
    g1[7] = 0u;
    asm volatile("tensor_load_to_lds %0, %1" :: "s"(g0), "s"(g1) : "memory");
}
#endif

// ---------------------------------------------------------------------------
__global__ __launch_bounds__(256)
void cvt_f16_kernel(const float* __restrict__ src, _Float16* __restrict__ dst, int n8)
{
    int i = blockIdx.x * blockDim.x + threadIdx.x;
    if (i < n8)
        *(half8*)(dst + (size_t)i * 8) = cvt8(src + (size_t)i * 8);
}

// W[K][N] fp32 -> Wt[N][K] f16 (transposed), 32x32 LDS tiles.
__global__ __launch_bounds__(256)
void transpose_w_kernel(const float* __restrict__ W, _Float16* __restrict__ Wt,
                        int K, int N)
{
    __shared__ float sT[32][33];
    const int tx = threadIdx.x, ty = threadIdx.y;
    const int n0 = blockIdx.x * 32, k0 = blockIdx.y * 32;
#pragma unroll
    for (int i = 0; i < 4; ++i)
        sT[ty + 8*i][tx] = W[(size_t)(k0 + ty + 8*i) * N + (n0 + tx)];
    __syncthreads();
#pragma unroll
    for (int i = 0; i < 4; ++i)
        Wt[(size_t)(n0 + ty + 8*i) * K + (k0 + tx)] = (_Float16)sT[tx][ty + 8*i];
}

// vh[(b*S+j)*DM + h*64 + d] f16 -> vt[((b*H+h)*64 + d)*S + j] f16.
// Grid: (S/32, 64/32, B*H); block (32,8). Coalesced both sides via LDS.
__global__ __launch_bounds__(256)
void transpose_v_kernel(const _Float16* __restrict__ vh, _Float16* __restrict__ vt,
                        int S, int DM, int H)
{
    __shared__ _Float16 sT[32][34];
    const int tx = threadIdx.x, ty = threadIdx.y;
    const int j0 = blockIdx.x * 32;
    const int d0 = blockIdx.y * 32;
    const int bh = blockIdx.z;
    const int b  = bh / H, h = bh - b * H;

    const _Float16* src = vh + ((size_t)b * S + j0) * DM + h * 64 + d0;
#pragma unroll
    for (int i = 0; i < 4; ++i)
        sT[ty + 8*i][tx] = src[(size_t)(ty + 8*i) * DM + tx];   // [j][d]
    __syncthreads();
    _Float16* dst = vt + ((size_t)bh * 64 + d0) * S + j0;
#pragma unroll
    for (int i = 0; i < 4; ++i)
        dst[(size_t)(ty + 8*i) * S + tx] = sT[tx][ty + 8*i];    // [d][j]
}

// ---------------------------------------------------------------------------
// GEMM: Y[M,N] = (X16[M,K] @ Wt[N,K]^T + bias[N]) * scale.
// 256 threads = 8 waves; 128x256 tile; 64x64 per wave; BK=32; TDM staging.
// ---------------------------------------------------------------------------
template <typename OutT>
__global__ __launch_bounds__(256)
void gemm_wmma_kernel(const _Float16* __restrict__ X, const _Float16* __restrict__ Wt,
                      const float* __restrict__ bias, OutT* __restrict__ Y,
                      int M, int N, int Kdim, float scale)
{
    __shared__ _Float16 sA[2][128][GEMM_LDSW];   // [m][k] padded rows
    __shared__ _Float16 sB[2][256][GEMM_LDSW];   // [n][k] padded rows

    const int tid   = threadIdx.x;
    const int lane  = tid & 31;
    const int wave  = tid >> 5;
    const int waveM = wave >> 2;          // 0..1
    const int waveN = wave & 3;           // 0..3
    const int blockM = blockIdx.y * 128;
    const int blockN = blockIdx.x * 256;

    const _Float16* gA = X  + (size_t)blockM * Kdim;
    const _Float16* gB = Wt + (size_t)blockN * Kdim;

#if USE_TDM
    const unsigned ldsA0  = (unsigned)(uintptr_t)&sA[0][0][0];
    const unsigned ldsB0  = (unsigned)(uintptr_t)&sB[0][0][0];
    const unsigned ldsAsz = 128u * GEMM_LDSW * 2u;
    const unsigned ldsBsz = 256u * GEMM_LDSW * 2u;
    // pad codes: 16 DWORDs (32 halves) -> interval code 3; 4 DWORDs pad -> code 3
    auto stage = [&](int k0, int buf) {
        if (wave == 0) {
            tdm_load_2d(gA + k0, ldsA0 + (unsigned)buf * ldsAsz,
                        (unsigned)Kdim, (unsigned)M, 32u, 128u, (unsigned)Kdim, 3u, 3u, true);
            tdm_load_2d(gB + k0, ldsB0 + (unsigned)buf * ldsBsz,
                        (unsigned)Kdim, (unsigned)N, 32u, 256u, (unsigned)Kdim, 3u, 3u, true);
        }
    };
    auto stage_wait = [&]() {
        if (wave == 0) __builtin_amdgcn_s_wait_tensorcnt(0);
    };
#else
    const _Float16* srcA0 = gA + (size_t)(tid >> 1) * Kdim + (tid & 1) * 16;
    const _Float16* srcB0 = gB + (size_t)tid * Kdim;
    auto stage = [&](int k0, int buf) {
        _Float16* da = &sA[buf][tid >> 1][(tid & 1) * 16];
        _Float16* db = &sB[buf][tid][0];
        *(half8*)(da)     = *(const half8*)(srcA0 + k0);
        *(half8*)(da + 8) = *(const half8*)(srcA0 + k0 + 8);
#pragma unroll
        for (int i = 0; i < 4; ++i)
            *(half8*)(db + i*8) = *(const half8*)(srcB0 + k0 + i*8);
    };
    auto stage_wait = [&]() {};
#endif

    const float8 zero8 = {0.f,0.f,0.f,0.f,0.f,0.f,0.f,0.f};
    float8 acc[4][4];
#pragma unroll
    for (int i = 0; i < 4; ++i)
#pragma unroll
        for (int j = 0; j < 4; ++j) acc[i][j] = zero8;

    const int nk = Kdim >> 5;
    stage(0, 0);
    stage_wait();
    __syncthreads();

    for (int kt = 0; kt < nk; ++kt) {
        const int cur = kt & 1;
        if (kt + 1 < nk) stage((kt + 1) << 5, cur ^ 1);

        half16 af[4], bf[4];
#pragma unroll
        for (int mt = 0; mt < 4; ++mt)
            af[mt] = load_frag(&sA[cur][waveM*64 + mt*16][0], GEMM_LDSW, lane);
#pragma unroll
        for (int nt = 0; nt < 4; ++nt)
            bf[nt] = load_frag(&sB[cur][waveN*64 + nt*16][0], GEMM_LDSW, lane);

#pragma unroll
        for (int mt = 0; mt < 4; ++mt)
#pragma unroll
            for (int nt = 0; nt < 4; ++nt)
                acc[mt][nt] = __builtin_amdgcn_wmma_f32_16x16x32_f16(
                    false, af[mt], false, bf[nt], (short)0, acc[mt][nt], false, false);

        stage_wait();
        __syncthreads();
    }

    const int rbase = blockM + waveM*64 + ((lane >> 4) << 3);
    const int cbase = blockN + waveN*64 + (lane & 15);
#pragma unroll
    for (int mt = 0; mt < 4; ++mt) {
        const int row0 = rbase + mt*16;
#pragma unroll
        for (int nt = 0; nt < 4; ++nt) {
            const int col = cbase + nt*16;
            const float bv = bias[col];
#pragma unroll
            for (int r = 0; r < 8; ++r) {
                float v = (acc[mt][nt][r] + bv) * scale;
                Y[(size_t)(row0 + r) * N + col] = (OutT)v;
            }
        }
    }
}

// ---------------------------------------------------------------------------
// Flash attention. Grid: (S/128, H, B). Block: 128 threads = 4 waves.
// K tiles [j][d] and pre-transposed V tiles [d][j], both TDM-staged and
// double-buffered. Scores arrive pre-scaled by log2e/sqrt(d) -> exp2 softmax.
// ---------------------------------------------------------------------------
__global__ __launch_bounds__(128)
void flash_attn_kernel(const _Float16* __restrict__ Qh,
                       const _Float16* __restrict__ Kh,
                       const _Float16* __restrict__ Vt,
                       _Float16* __restrict__ O,
                       int S, int DM, int H)
{
    __shared__ _Float16 sK[2][64][ATTN_LDSW];    // [j][d] padded
    __shared__ _Float16 sVt[2][64][ATTN_LDSW];   // [d][j] padded
    __shared__ _Float16 sP[128][ATTN_LDSW];      // [q][j] C->A layout bounce

    const int tid  = threadIdx.x;
    const int lane = tid & 31;
    const int wave = tid >> 5;
    const int qblk = blockIdx.x;
    const int h    = blockIdx.y;
    const int b    = blockIdx.z;

    const size_t rowStride = (size_t)DM;
    const size_t headOff   = (size_t)h * 64;
    const size_t batchOff  = (size_t)b * S * rowStride;
    const int qrow0 = qblk * 128 + wave * 32;

    half16 qf[2][2];
#pragma unroll
    for (int mt = 0; mt < 2; ++mt)
#pragma unroll
        for (int ks = 0; ks < 2; ++ks) {
            const int r  = lane & 15;
            const int kh = ((lane >> 4) << 3) + ks * 32;
            const _Float16* p = Qh + batchOff + (size_t)(qrow0 + mt*16 + r) * rowStride + headOff + kh;
            half8 lo = *(const half8*)(p);
            half8 hi = *(const half8*)(p + 16);
            qf[mt][ks] = __builtin_shufflevector(lo, hi, 0,1,2,3,4,5,6,7,8,9,10,11,12,13,14,15);
        }

    const float8 zero8 = {0.f,0.f,0.f,0.f,0.f,0.f,0.f,0.f};
    float8 of[2][4];
#pragma unroll
    for (int mt = 0; mt < 2; ++mt)
#pragma unroll
        for (int nt = 0; nt < 4; ++nt) of[mt][nt] = zero8;

    float8 mrow[2], lrow[2];
#pragma unroll
    for (int mt = 0; mt < 2; ++mt)
#pragma unroll
        for (int e = 0; e < 8; ++e) { mrow[mt][e] = -3.0e38f; lrow[mt][e] = 0.f; }

    const _Float16* gK  = Kh + batchOff + headOff;               // rows j, 64 cols
    const _Float16* gVt = Vt + (size_t)(b * H + h) * 64 * S;     // rows d, S cols

#if USE_TDM
    const unsigned ldsK0  = (unsigned)(uintptr_t)&sK[0][0][0];
    const unsigned ldsV0  = (unsigned)(uintptr_t)&sVt[0][0][0];
    const unsigned ldsTsz = 64u * ATTN_LDSW * 2u;
    // pad codes: 32 DWORDs (64 halves) -> interval code 4; 4 DWORDs pad -> code 3
    auto stageKV = [&](int j0, int buf) {
        if (wave == 0) {
            tdm_load_2d(gK + (size_t)j0 * rowStride, ldsK0 + (unsigned)buf * ldsTsz,
                        (unsigned)DM, (unsigned)S, 64u, 64u, (unsigned)DM, 4u, 3u, true);
            tdm_load_2d(gVt + j0, ldsV0 + (unsigned)buf * ldsTsz,
                        (unsigned)S, 64u, 64u, 64u, (unsigned)S, 4u, 3u, true);
        }
    };
    auto stage_wait = [&]() {
        if (wave == 0) __builtin_amdgcn_s_wait_tensorcnt(0);
    };
#else
    const int krow = tid >> 1, khf = tid & 1;
    auto stageKV = [&](int j0, int buf) {
        const _Float16* srcK = gK + (size_t)(j0 + krow) * rowStride + khf*32;
        _Float16* dstK = &sK[buf][krow][khf*32];
#pragma unroll
        for (int i = 0; i < 4; ++i)
            *(half8*)(dstK + i*8) = *(const half8*)(srcK + i*8);
        const _Float16* srcV = gVt + (size_t)krow * S + j0 + khf*32;
        _Float16* dstV = &sVt[buf][krow][khf*32];
#pragma unroll
        for (int i = 0; i < 4; ++i)
            *(half8*)(dstV + i*8) = *(const half8*)(srcV + i*8);
    };
    auto stage_wait = [&]() {};
#endif

    const int nj = S >> 6;
    stageKV(0, 0);
    stage_wait();
    __syncthreads();

    for (int jt = 0; jt < nj; ++jt) {
        const int cur = jt & 1;
        if (jt + 1 < nj) stageKV((jt + 1) << 6, cur ^ 1);

        // ---- scores S = q @ K^T (already scaled by log2e/sqrt(d))
        float8 sf[2][4];
#pragma unroll
        for (int mt = 0; mt < 2; ++mt)
#pragma unroll
            for (int nt = 0; nt < 4; ++nt) sf[mt][nt] = zero8;
#pragma unroll
        for (int ks = 0; ks < 2; ++ks) {
            half16 bf[4];
#pragma unroll
            for (int nt = 0; nt < 4; ++nt)
                bf[nt] = load_frag(&sK[cur][nt*16][ks*32], ATTN_LDSW, lane);
#pragma unroll
            for (int mt = 0; mt < 2; ++mt)
#pragma unroll
                for (int nt = 0; nt < 4; ++nt)
                    sf[mt][nt] = __builtin_amdgcn_wmma_f32_16x16x32_f16(
                        false, qf[mt][ks], false, bf[nt], (short)0, sf[mt][nt], false, false);
        }

        // ---- online softmax in exp2 domain (C layout: VGPR r = row)
#pragma unroll
        for (int mt = 0; mt < 2; ++mt) {
            float8 rm = sf[mt][0];
#pragma unroll
            for (int nt = 1; nt < 4; ++nt)
#pragma unroll
                for (int e = 0; e < 8; ++e) rm[e] = fmaxf(rm[e], sf[mt][nt][e]);
#pragma unroll
            for (int off = 8; off > 0; off >>= 1)
#pragma unroll
                for (int e = 0; e < 8; ++e)
                    rm[e] = fmaxf(rm[e], __shfl_xor(rm[e], off, 16));

            float8 mn, alpha;
#pragma unroll
            for (int e = 0; e < 8; ++e) {
                mn[e]    = fmaxf(mrow[mt][e], rm[e]);
                alpha[e] = exp2f(mrow[mt][e] - mn[e]);
                mrow[mt][e] = mn[e];
                lrow[mt][e] *= alpha[e];
            }
#pragma unroll
            for (int nt = 0; nt < 4; ++nt)
#pragma unroll
                for (int e = 0; e < 8; ++e) of[mt][nt][e] *= alpha[e];

            float8 rs = zero8;
#pragma unroll
            for (int nt = 0; nt < 4; ++nt)
#pragma unroll
                for (int e = 0; e < 8; ++e) {
                    float p = exp2f(sf[mt][nt][e] - mn[e]);
                    sf[mt][nt][e] = p;
                    rs[e] += p;
                }
#pragma unroll
            for (int off = 8; off > 0; off >>= 1)
#pragma unroll
                for (int e = 0; e < 8; ++e)
                    rs[e] += __shfl_xor(rs[e], off, 16);
#pragma unroll
            for (int e = 0; e < 8; ++e) lrow[mt][e] += rs[e];

            // ---- store P (C-layout -> A-layout via wave-private strip)
            const int prow0 = wave*32 + mt*16 + ((lane >> 4) << 3);
#pragma unroll
            for (int nt = 0; nt < 4; ++nt) {
                const int col = nt*16 + (lane & 15);
#pragma unroll
                for (int r = 0; r < 8; ++r)
                    sP[prow0 + r][col] = (_Float16)sf[mt][nt][r];
            }
        }

        // ---- O += P @ V (wave-private strips; in-wave LDS ordering suffices)
#pragma unroll
        for (int ks = 0; ks < 2; ++ks) {
            half16 pf[2], vf[4];
#pragma unroll
            for (int mt = 0; mt < 2; ++mt)
                pf[mt] = load_frag(&sP[wave*32 + mt*16][ks*32], ATTN_LDSW, lane);
#pragma unroll
            for (int nt = 0; nt < 4; ++nt)
                vf[nt] = load_frag(&sVt[cur][nt*16][ks*32], ATTN_LDSW, lane);
#pragma unroll
            for (int mt = 0; mt < 2; ++mt)
#pragma unroll
                for (int nt = 0; nt < 4; ++nt)
                    of[mt][nt] = __builtin_amdgcn_wmma_f32_16x16x32_f16(
                        false, pf[mt], false, vf[nt], (short)0, of[mt][nt], false, false);
        }

        stage_wait();
        __syncthreads();
    }

    // ---- finalize
#pragma unroll
    for (int mt = 0; mt < 2; ++mt) {
        float8 inv;
#pragma unroll
        for (int e = 0; e < 8; ++e) inv[e] = 1.0f / lrow[mt][e];
        const int row0 = qrow0 + mt*16 + ((lane >> 4) << 3);
#pragma unroll
        for (int nt = 0; nt < 4; ++nt) {
            const int col = (int)headOff + nt*16 + (lane & 15);
#pragma unroll
            for (int r = 0; r < 8; ++r)
                O[batchOff + (size_t)(row0 + r) * rowStride + col] =
                    (_Float16)(of[mt][nt][r] * inv[r]);
        }
    }
}

// ---------------------------------------------------------------------------
extern "C" void kernel_launch(void* const* d_in, const int* in_sizes, int n_in,
                              void* d_out, int out_size, void* d_ws, size_t ws_size,
                              hipStream_t stream) {
    (void)in_sizes; (void)n_in; (void)out_size; (void)ws_size;

    const float* Q  = (const float*)d_in[0];
    const float* K  = (const float*)d_in[1];
    const float* V  = (const float*)d_in[2];
    const float* Wq = (const float*)d_in[3];
    const float* bq = (const float*)d_in[4];
    const float* Wk = (const float*)d_in[5];
    const float* bk = (const float*)d_in[6];
    const float* Wv = (const float*)d_in[7];
    const float* bv = (const float*)d_in[8];
    const float* Wo = (const float*)d_in[9];
    const float* bo = (const float*)d_in[10];

    const int B = 2, S = 2048, DM = 1024, H = 16;
    const int M = B * S;
    // 1/sqrt(64) * log2(e): scores land pre-scaled for exp2 softmax
    const float qScale = 0.125f * 1.44269504f;
    const size_t MD = (size_t)M * DM;
    const size_t DD = (size_t)DM * DM;

    _Float16* x16q = (_Float16*)d_ws;
    _Float16* x16k = x16q + MD;
    _Float16* x16v = x16k + MD;
    _Float16* wtq  = x16v + MD;
    _Float16* wtk  = wtq + DD;
    _Float16* wtv  = wtk + DD;
    _Float16* wto  = wtv + DD;
    _Float16* qh   = wto + DD;
    _Float16* kh   = qh + MD;
    _Float16* vh   = kh + MD;
    _Float16* vt   = vh + MD;      // [B][H][64][S]
    _Float16* att  = vt + MD;

    {
        const int n8 = (int)(MD / 8);
        dim3 blk(256), grd((n8 + 255) / 256);
        cvt_f16_kernel<<<grd, blk, 0, stream>>>(Q, x16q, n8);
        cvt_f16_kernel<<<grd, blk, 0, stream>>>(K, x16k, n8);
        cvt_f16_kernel<<<grd, blk, 0, stream>>>(V, x16v, n8);
    }
    {
        dim3 blk(32, 8), grd(DM / 32, DM / 32);
        transpose_w_kernel<<<grd, blk, 0, stream>>>(Wq, wtq, DM, DM);
        transpose_w_kernel<<<grd, blk, 0, stream>>>(Wk, wtk, DM, DM);
        transpose_w_kernel<<<grd, blk, 0, stream>>>(Wv, wtv, DM, DM);
        transpose_w_kernel<<<grd, blk, 0, stream>>>(Wo, wto, DM, DM);
    }

    dim3 blk(256);
    dim3 grid_gemm(DM / 256, M / 128);
    gemm_wmma_kernel<_Float16><<<grid_gemm, blk, 0, stream>>>(x16q, wtq, bq, qh, M, DM, DM, qScale);
    gemm_wmma_kernel<_Float16><<<grid_gemm, blk, 0, stream>>>(x16k, wtk, bk, kh, M, DM, DM, 1.0f);
    gemm_wmma_kernel<_Float16><<<grid_gemm, blk, 0, stream>>>(x16v, wtv, bv, vh, M, DM, DM, 1.0f);

    {   // one-time V transpose to [B][H][d][S]
        dim3 tblk(32, 8), tgrd(S / 32, 2, B * H);
        transpose_v_kernel<<<tgrd, tblk, 0, stream>>>(vh, vt, S, DM, H);
    }

    dim3 grid_attn(S / 128, H, B);
    flash_attn_kernel<<<grid_attn, dim3(128), 0, stream>>>(qh, kh, vt, att, S, DM, H);

    gemm_wmma_kernel<float><<<grid_gemm, blk, 0, stream>>>(att, wto, bo, (float*)d_out, M, DM, DM, 1.0f);
}
